// SelfAttentionGMM_14508399526601
// MI455X (gfx1250) — compile-verified, matrix operand order checked by reading, not run on hardware
//
#include <hip/hip_runtime.h>
#include <hip/hip_bf16.h>

// ---------------- constants ----------------
#define BB   32
#define NN   1024
#define DD   64
#define NSLOT 8
#define HIDN 128
#define HEADS 4
#define DH   16
#define MM   1026          // 2 + N tokens
#define MP   1040          // padded to 65*16
#define MT   65            // MP/16
#define EPSV 1e-8f
#define SCALEV 0.125f      // 64^-0.5
#define TSCALEV 0.25f      // 16^-0.5

typedef _Float16 h16;
typedef __attribute__((ext_vector_type(16))) _Float16 v16h;
typedef __attribute__((ext_vector_type(8)))  float    v8f;

// ---------------- WMMA helpers (ISA 7.12.2 layouts, wave32) ----------------
__device__ __forceinline__ int lane32() { return (int)(threadIdx.x & 31u); }

__device__ __forceinline__ v8f wmma16(v16h a, v16h b, v8f c) {
  return __builtin_amdgcn_wmma_f32_16x16x32_f16(false, a, false, b, (short)0, c,
                                                false, false);
}

// A 16x32 f16: lanes 0-15 M=lane K=0..7/16..23, lanes 16-31 M=lane-16 K=8..15/24..31
template <typename F>
__device__ __forceinline__ v16h frag_a(F f) {
  int l = lane32();
  int m = l & 15;
  int kb = (l & 16) ? 8 : 0;
  v16h r;
#pragma unroll
  for (int e = 0; e < 16; ++e) {
    int k = kb + (e < 8 ? e : e + 8);
    r[e] = f(m, k);
  }
  return r;
}
// B 32x16 f16 (K x N): mirror of A with N striped across lanes
template <typename F>
__device__ __forceinline__ v16h frag_b(F f) {
  int l = lane32();
  int n = l & 15;
  int kb = (l & 16) ? 8 : 0;
  v16h r;
#pragma unroll
  for (int e = 0; e < 16; ++e) {
    int k = kb + (e < 8 ? e : e + 8);
    r[e] = f(k, n);
  }
  return r;
}
// C/D 16x16 f32: lane = N (0..15 twice), vgpr r -> M = r + 8*(lane>=16)
template <typename F>
__device__ __forceinline__ void store_c(const v8f& c, F f) {
  int l = lane32();
  int n = l & 15;
  int mb = (l & 16) ? 8 : 0;
#pragma unroll
  for (int r = 0; r < 8; ++r) f(mb + r, n, c[r]);
}

// ---------------- kernel: build x0 = [mu_p; logsigma_p; inputs; 0-pad] ----
__global__ void k_build_x(const float* __restrict__ inputs,
                          const float* __restrict__ mu_p,
                          const float* __restrict__ ls_p,
                          float* __restrict__ X) {
  long i = (long)blockIdx.x * 256 + threadIdx.x;
  if (i >= (long)BB * MP * DD) return;
  int d = (int)(i % DD);
  long t = i / DD;
  int m = (int)(t % MP);
  int b = (int)(t / MP);
  float v;
  if (m == 0)       v = mu_p[d];
  else if (m == 1)  v = ls_p[d];
  else if (m < MM)  v = inputs[((long)b * NN + (m - 2)) * DD + d];
  else              v = 0.0f;
  X[i] = v;
}

// ---------------- kernel: row LayerNorm -> f16 ---------------------------
__global__ void k_ln(const float* __restrict__ Xin, long bsIn, int row0, int nrows,
                     const float* __restrict__ g, const float* __restrict__ bt,
                     h16* __restrict__ out16, long bsOut, int dim) {
  int r = blockIdx.x * blockDim.x + threadIdx.x;
  if (r >= BB * nrows) return;
  int b = r / nrows, m = r % nrows;
  const float* x = Xin + (long)b * bsIn + (long)(row0 + m) * dim;
  float mean = 0.f;
  for (int i = 0; i < dim; ++i) mean += x[i];
  mean /= (float)dim;
  float var = 0.f;
  for (int i = 0; i < dim; ++i) { float d = x[i] - mean; var += d * d; }
  var /= (float)dim;
  float inv = rsqrtf(var + 1e-5f);
  h16* o = out16 + (long)b * bsOut + (long)m * dim;
  for (int i = 0; i < dim; ++i) o[i] = (h16)((x[i] - mean) * inv * g[i] + bt[i]);
}

// ---------------- kernel: generic WMMA GEMM  C = act(A @ W^T + bias) (+res)
// one wave (32 thr) per 16x16 tile.  A: f16 [rows x K] row-major per batch,
// W: f32 [N x K] row-major shared.  BIAS: add bias[n]. ACT: 0 none,1 gelu,2 relu.
// RES: add/readback C32 in place.
template <int BIAS, int ACT, int RES>
__global__ void k_gemm(const h16* __restrict__ A, long bsA, int lda,
                       const float* __restrict__ W, int K,
                       const float* __restrict__ bias,
                       float* C32, long bsC32, int ldc32,
                       h16* C16, long bsC16, int ldc16) {
  int mt = blockIdx.x, nt = blockIdx.y, b = blockIdx.z;
  const h16* Ab = A + (long)b * bsA + (long)mt * 16 * lda;
  __builtin_prefetch(&W[(long)(nt * 16) * K], 0, 0);
  v8f acc = {};
  for (int kc = 0; kc < K; kc += 32) {
    v16h af = frag_a([&](int m, int k) { return Ab[(long)m * lda + kc + k]; });
    v16h bf = frag_b([&](int k, int n) {
      return (h16)W[(long)(nt * 16 + n) * K + kc + k];
    });
    acc = wmma16(af, bf, acc);
  }
  store_c(acc, [&](int m, int n, float v) {
    int gm = mt * 16 + m, gn = nt * 16 + n;
    float x = v;
    if (BIAS) x += bias[gn];
    if (ACT == 1) {  // tanh-approx gelu (jax default)
      float t = 0.7978845608f * (x + 0.044715f * x * x * x);
      x = 0.5f * x * (1.0f + tanhf(t));
    } else if (ACT == 2) {
      x = fmaxf(x, 0.0f);
    }
    if (RES) x += C32[(long)b * bsC32 + (long)gm * ldc32 + gn];
    if (C32) C32[(long)b * bsC32 + (long)gm * ldc32 + gn] = x;
    if (C16) C16[(long)b * bsC16 + (long)gm * ldc16 + gn] = (h16)x;
  });
}

// ---------------- kernel: MHA, one workgroup per (batch, head) ------------
// QKV f16 [B][MP][192] (q:0..63 k:64..127 v:128..191), O16 f16 [B][MP][64]
__global__ __launch_bounds__(256) void k_attn(const h16* __restrict__ QKV,
                                              h16* __restrict__ O16) {
  extern __shared__ char smem[];
  h16*   Ksh   = (h16*)smem;                 // [MP][DH]
  h16*   Vsh   = Ksh + MP * DH;              // [MP][DH]
  float* Ssh   = (float*)(Vsh + MP * DH);    // [16][MP]
  float* Opart = Ssh + 16 * MP;              // [8][16][16]

  int b = blockIdx.x / HEADS, hd = blockIdx.x % HEADS;
  int tid = threadIdx.x, wave = tid >> 5;
  const h16* base = QKV + (long)b * MP * 192;

  for (int i = tid; i < MP * DH; i += 256) {
    int row = i / DH, d = i % DH;
    Ksh[i] = base[(long)row * 192 + 64 + hd * 16 + d];
    Vsh[i] = base[(long)row * 192 + 128 + hd * 16 + d];
  }
  __syncthreads();

  for (int qt = 0; qt < MT; ++qt) {
    const h16* Qb = base + (long)qt * 16 * 192 + hd * 16;
    // scores: S[m][n] = sum_d Q[m][d]*K[n][d]   (K-dim = 16 padded to 32)
    for (int jt = wave; jt < MT; jt += 8) {
      v8f acc = {};
      v16h af = frag_a([&](int m, int k) {
        return (k < DH) ? Qb[(long)m * 192 + k] : (h16)0.0f;
      });
      v16h bf = frag_b([&](int k, int n) {
        return (k < DH) ? Ksh[(jt * 16 + n) * DH + k] : (h16)0.0f;
      });
      acc = wmma16(af, bf, acc);
      store_c(acc, [&](int m, int n, float v) { Ssh[m * MP + jt * 16 + n] = v; });
    }
    __syncthreads();
    // softmax over 1026 valid keys, 16 lanes per row
    {
      int row = tid >> 4, c0 = tid & 15;
      float mx = -3.4e38f;
      for (int n = c0; n < MP; n += 16) {
        float s = (n < MM) ? Ssh[row * MP + n] * TSCALEV : -3.4e38f;
        Ssh[row * MP + n] = s;
        mx = fmaxf(mx, s);
      }
      for (int o = 8; o; o >>= 1) mx = fmaxf(mx, __shfl_xor(mx, o, 16));
      float sum = 0.f;
      for (int n = c0; n < MP; n += 16) {
        float e = (n < MM) ? __expf(Ssh[row * MP + n] - mx) : 0.0f;
        Ssh[row * MP + n] = e;
        sum += e;
      }
      for (int o = 8; o; o >>= 1) sum += __shfl_xor(sum, o, 16);
      float inv = 1.0f / sum;
      for (int n = c0; n < MP; n += 16) Ssh[row * MP + n] *= inv;
    }
    __syncthreads();
    // O = P @ V, 33 K-chunks of 32 split across 8 waves, then LDS reduce
    v8f acc = {};
    for (int kc = wave; kc < 33; kc += 8) {
      int k0 = kc * 32;
      v16h af = frag_a([&](int m, int k) {
        int kk = k0 + k;
        return (kk < MP) ? (h16)Ssh[m * MP + kk] : (h16)0.0f;
      });
      v16h bf = frag_b([&](int k, int n) {
        int kk = k0 + k;
        return (kk < MP) ? Vsh[kk * DH + n] : (h16)0.0f;
      });
      acc = wmma16(af, bf, acc);
    }
    store_c(acc, [&](int m, int n, float v) { Opart[wave * 256 + m * 16 + n] = v; });
    __syncthreads();
    {
      int m = tid >> 4, n = tid & 15;
      float s = 0.f;
      for (int w = 0; w < 8; ++w) s += Opart[w * 256 + m * 16 + n];
      O16[(long)b * MP * 64 + (long)(qt * 16 + m) * 64 + hd * 16 + n] = (h16)s;
    }
    __syncthreads();
  }
}

// ---------------- kernel: slot init (mu/logsigma MLPs + reparam) ----------
__global__ __launch_bounds__(256) void k_slots_init(
    const float* __restrict__ X, const float* __restrict__ noise,
    const float* miW1, const float* mib1, const float* miW2, const float* mib2,
    const float* siW1, const float* sib1, const float* siW2, const float* sib2,
    float* SLOTS, float* SLOTS0, float* PI) {
  __shared__ float mt[64], lt[64], h1[128], h2[128], mu[128], ls[128];
  int b = blockIdx.x, t = threadIdx.x;
  if (t < 64) {
    mt[t] = X[((long)b * MP + 0) * 64 + t];
    lt[t] = X[((long)b * MP + 1) * 64 + t];
  }
  __syncthreads();
  if (t < 128) {
    float a = mib1[t], c = sib1[t];
    for (int k = 0; k < 64; ++k) {
      a += mt[k] * miW1[t * 64 + k];
      c += lt[k] * siW1[t * 64 + k];
    }
    h1[t] = fmaxf(a, 0.f);
    h2[t] = fmaxf(c, 0.f);
  }
  __syncthreads();
  if (t < 128) {
    float a = mib2[t], c = sib2[t];
    for (int k = 0; k < 128; ++k) {
      a += h1[k] * miW2[t * 128 + k];
      c += h2[k] * siW2[t * 128 + k];
    }
    mu[t] = a;
    ls[t] = c;
  }
  __syncthreads();
  for (int i = t; i < NSLOT * 128; i += 256) {
    int c = i & 127;
    float v = mu[c] + __expf(ls[c]) * noise[(long)b * NSLOT * 128 + i];
    SLOTS[(long)b * NSLOT * 128 + i] = v;
    SLOTS0[(long)b * NSLOT * 128 + i] = v;
  }
  if (t < NSLOT) PI[b * NSLOT + t] = 1.0f / NSLOT;
}

// ---------------- kernel: one slot-attention GMM step, 1 WG per batch -----
__global__ __launch_bounds__(256) void k_step(
    float* SLOTS, float* PI, const float* __restrict__ Kf,
    const float* __restrict__ Vf, const h16* __restrict__ V16,
    const float* Wq, const float* gWih, const float* gWhh, const float* gbih,
    const float* gbhh, const float* mW1, const float* mb1, const float* mW2,
    const float* mb2, const float* lnslg, const float* lnslb,
    const float* lnmug, const float* lnmub, float* out_attn) {
  extern __shared__ char smem[];
  float* sl    = (float*)smem;         // 8*128
  float* sln   = sl + NSLOT * 128;     // 8*128
  float* qmu   = sln + NSLOT * 128;    // 8*64
  float* ivar  = qmu + NSLOT * 64;     // 8*64
  float* att   = ivar + NSLOT * 64;    // 8*1024
  float* xg    = att + NSLOT * 1024;   // 8*192
  float* hg    = xg + NSLOT * 192;     // 8*192
  float* updmu = hg + NSLOT * 192;     // 16*64
  float* ln64  = updmu + 16 * 64;      // 8*64
  float* hid   = ln64 + NSLOT * 64;    // 8*128
  float* updls = hid + NSLOT * 128;    // 8*64
  float* rs    = updls + NSLOT * 64;   // 8
  float* pn    = rs + NSLOT;           // 8
  h16* att16   = (h16*)(pn + NSLOT);   // 16*1024

  int b = blockIdx.x, t = threadIdx.x, wave = t >> 5;
  const float* Kb = Kf + (long)b * NN * DD;
  const float* Vb = Vf + (long)b * NN * DD;
  const h16* V16b = V16 + (long)b * NN * DD;

  for (int i = t; i < NSLOT * 128; i += 256) sl[i] = SLOTS[(long)b * NSLOT * 128 + i];
  __syncthreads();
  // LN(slots) over 128
  if (t < NSLOT) {
    float mean = 0.f;
    for (int i = 0; i < 128; ++i) mean += sl[t * 128 + i];
    mean /= 128.f;
    float var = 0.f;
    for (int i = 0; i < 128; ++i) { float d = sl[t * 128 + i] - mean; var += d * d; }
    var /= 128.f;
    float inv = rsqrtf(var + 1e-5f);
    for (int i = 0; i < 128; ++i)
      sln[t * 128 + i] = (sl[t * 128 + i] - mean) * inv * lnslg[i] + lnslb[i];
  }
  __syncthreads();
  // q_mu = s_mu @ Wq^T ; inv_var = exp(-2 * s_ls @ Wq^T)
  for (int idx = t; idx < 1024; idx += 256) {
    int s = (idx >> 6) & 7, d = idx & 63;
    bool isls = idx >= 512;
    const float* srow = sln + s * 128 + (isls ? 64 : 0);
    float a = 0.f;
    for (int k = 0; k < 64; ++k) a += srow[k] * Wq[d * 64 + k];
    if (isls) ivar[s * 64 + d] = __expf(-2.0f * a);
    else      qmu[s * 64 + d] = a;
  }
  __syncthreads();
  // dots_exp
  for (int idx = t; idx < NSLOT * 1024; idx += 256) {
    int s = idx >> 10, n = idx & 1023;
    float acc = 0.f;
    for (int d = 0; d < 64; ++d) {
      float df = Kb[(long)n * 64 + d] - qmu[s * 64 + d];
      acc += df * df * ivar[s * 64 + d];
    }
    att[idx] = (__expf(-acc * SCALEV) + EPSV) * PI[b * NSLOT + s];
  }
  __syncthreads();
  // normalize over slots, then over n
  for (int n = t; n < 1024; n += 256) {
    float cs = 0.f;
    for (int s = 0; s < NSLOT; ++s) cs += att[s * 1024 + n];
    float inv = 1.0f / cs;
    for (int s = 0; s < NSLOT; ++s) att[s * 1024 + n] *= inv;
  }
  __syncthreads();
  if (t < NSLOT) {
    float r2 = 0.f;
    for (int n = 0; n < 1024; ++n) r2 += att[t * 1024 + n];
    rs[t] = r2;
  }
  __syncthreads();
  for (int idx = t; idx < NSLOT * 1024; idx += 256) att[idx] /= rs[idx >> 10];
  __syncthreads();
  if (t < NSLOT) {
    float r2 = 0.f;
    for (int n = 0; n < 1024; ++n) r2 += att[t * 1024 + n];
    pn[t] = r2;
  }
  __syncthreads();
  if (t == 0) {
    float tot = EPSV;
    for (int s = 0; s < NSLOT; ++s) tot += pn[s];
    for (int s = 0; s < NSLOT; ++s) PI[b * NSLOT + s] = pn[s] / tot;
  }
  // pack attention to f16 (rows 8..15 zero) for WMMA
  for (int i = t; i < 16 * 1024; i += 256)
    att16[i] = (i >> 10) < NSLOT ? (h16)att[i] : (h16)0.0f;
  __syncthreads();
  // upd_mu = attn @ V : 16x64 out, K = 1024 (32 chunks), 4 waves x 16-col tiles
  if (wave < 4) {
    v8f acc = {};
    for (int kc = 0; kc < 32; ++kc) {
      v16h af = frag_a([&](int m, int k) { return att16[m * 1024 + kc * 32 + k]; });
      v16h bf = frag_b([&](int k, int n) {
        return V16b[(long)(kc * 32 + k) * 64 + wave * 16 + n];
      });
      acc = wmma16(af, bf, acc);
    }
    store_c(acc, [&](int m, int n, float v) { updmu[m * 64 + wave * 16 + n] = v; });
  }
  __syncthreads();
  // GRU gates
  for (int idx = t; idx < NSLOT * 192; idx += 256) {
    int s = idx / 192, j = idx % 192;
    float a = gbih[j], c = gbhh[j];
    for (int k = 0; k < 64; ++k) {
      a += updmu[s * 64 + k] * gWih[j * 64 + k];
      c += sln[s * 128 + k] * gWhh[j * 64 + k];
    }
    xg[idx] = a;
    hg[idx] = c;
  }
  __syncthreads();
  for (int idx = t; idx < NSLOT * 64; idx += 256) {
    int s = idx >> 6, dd = idx & 63;
    float r = 1.f / (1.f + __expf(-(xg[s * 192 + dd] + hg[s * 192 + dd])));
    float z = 1.f / (1.f + __expf(-(xg[s * 192 + 64 + dd] + hg[s * 192 + 64 + dd])));
    float cand = tanhf(xg[s * 192 + 128 + dd] + r * hg[s * 192 + 128 + dd]);
    updls[idx] = (1.f - z) * cand + z * sln[s * 128 + dd];  // temp: hnew
  }
  __syncthreads();
  for (int idx = t; idx < NSLOT * 64; idx += 256) updmu[idx] = updls[idx];
  __syncthreads();
  // residual MLP on mu
  if (t < NSLOT) {
    float mean = 0.f;
    for (int i = 0; i < 64; ++i) mean += updmu[t * 64 + i];
    mean /= 64.f;
    float var = 0.f;
    for (int i = 0; i < 64; ++i) { float d = updmu[t * 64 + i] - mean; var += d * d; }
    var /= 64.f;
    float inv = rsqrtf(var + 1e-5f);
    for (int i = 0; i < 64; ++i)
      ln64[t * 64 + i] = (updmu[t * 64 + i] - mean) * inv * lnmug[i] + lnmub[i];
  }
  __syncthreads();
  for (int idx = t; idx < NSLOT * 128; idx += 256) {
    int s = idx >> 7, h = idx & 127;
    float a = mb1[h];
    for (int k = 0; k < 64; ++k) a += ln64[s * 64 + k] * mW1[h * 64 + k];
    hid[idx] = fmaxf(a, 0.f);
  }
  __syncthreads();
  for (int idx = t; idx < NSLOT * 64; idx += 256) {
    int s = idx >> 6, dd = idx & 63;
    float a = mb2[dd];
    for (int k = 0; k < 128; ++k) a += hid[s * 128 + k] * mW2[dd * 128 + k];
    updmu[idx] += a;
  }
  __syncthreads();
  // upd_ls = 0.5*log(sum_n ((v-mu)^2+eps)*attn)
  for (int idx = t; idx < NSLOT * 64; idx += 256) {
    int s = idx >> 6, dd = idx & 63;
    float um = updmu[idx], acc = 0.f;
    for (int n = 0; n < 1024; ++n) {
      float df = Vb[(long)n * 64 + dd] - um;
      acc += (df * df + EPSV) * att[s * 1024 + n];
    }
    updls[idx] = 0.5f * __logf(acc);
  }
  __syncthreads();
  for (int idx = t; idx < NSLOT * 128; idx += 256) {
    int s = idx >> 7, c = idx & 127;
    SLOTS[(long)b * NSLOT * 128 + idx] =
        (c < 64) ? updmu[s * 64 + c] : updls[s * 64 + (c - 64)];
  }
  for (int idx = t; idx < NSLOT * 1024; idx += 256)
    out_attn[(long)b * NSLOT * 1024 + idx] = att[idx];
}

// ---------------- kernel: output MLP, one WG per (b,slot) ----------------
__global__ __launch_bounds__(256) void k_mlp_out(const float* __restrict__ SL,
                                                 const float* W1, const float* b1,
                                                 const float* W2, const float* b2,
                                                 float* __restrict__ outp) {
  __shared__ float srow[128], hid[256];
  int row = blockIdx.x, t = threadIdx.x;
  for (int i = t; i < 128; i += 256) srow[i] = SL[(long)row * 128 + i];
  __syncthreads();
  {
    float a = b1[t];
    for (int k = 0; k < 128; ++k) a += srow[k] * W1[t * 128 + k];
    hid[t] = fmaxf(a, 0.f);
  }
  __syncthreads();
  if (t < 64) {
    float a = b2[t];
    for (int k = 0; k < 256; ++k) a += hid[k] * W2[t * 256 + k];
    outp[(long)row * 64 + t] = a;
  }
}

// ---------------- host launcher -------------------------------------------
extern "C" void kernel_launch(void* const* d_in, const int* in_sizes, int n_in,
                              void* d_out, int out_size, void* d_ws, size_t ws_size,
                              hipStream_t stream) {
  const float* inputs = (const float*)d_in[0];
  const float* noise  = (const float*)d_in[1];
  const float* mu_p   = (const float*)d_in[2];
  const float* ls_p   = (const float*)d_in[3];
  const float* Wq     = (const float*)d_in[4];
  const float* Wk     = (const float*)d_in[5];
  const float* Wv     = (const float*)d_in[6];
  const float* gWih   = (const float*)d_in[7];
  const float* gWhh   = (const float*)d_in[8];
  const float* gbih   = (const float*)d_in[9];
  const float* gbhh   = (const float*)d_in[10];
  const float* mW1    = (const float*)d_in[11];
  const float* mb1    = (const float*)d_in[12];
  const float* mW2    = (const float*)d_in[13];
  const float* mb2    = (const float*)d_in[14];
  const float* ln_in_g = (const float*)d_in[15];
  const float* ln_in_b = (const float*)d_in[16];
  const float* ln_sl_g = (const float*)d_in[17];
  const float* ln_sl_b = (const float*)d_in[18];
  const float* ln_mu_g = (const float*)d_in[19];
  const float* ln_mu_b = (const float*)d_in[20];
  const float* oW1    = (const float*)d_in[21];
  const float* ob1    = (const float*)d_in[22];
  const float* oW2    = (const float*)d_in[23];
  const float* ob2    = (const float*)d_in[24];
  const float* miW1   = (const float*)d_in[25];
  const float* mib1   = (const float*)d_in[26];
  const float* miW2   = (const float*)d_in[27];
  const float* mib2   = (const float*)d_in[28];
  const float* siW1   = (const float*)d_in[29];
  const float* sib1   = (const float*)d_in[30];
  const float* siW2   = (const float*)d_in[31];
  const float* sib2   = (const float*)d_in[32];
  const float* t_ln1_g = (const float*)d_in[33];
  const float* t_ln1_b = (const float*)d_in[34];
  const float* t_Wqkv  = (const float*)d_in[35];
  const float* t_Wo    = (const float*)d_in[36];
  const float* t_bo    = (const float*)d_in[37];
  const float* t_ln2_g = (const float*)d_in[38];
  const float* t_ln2_b = (const float*)d_in[39];
  const float* t_W1    = (const float*)d_in[40];
  const float* t_b1    = (const float*)d_in[41];
  const float* t_W2    = (const float*)d_in[42];
  const float* t_b2    = (const float*)d_in[43];

  // workspace carve
  char* w = (char*)d_ws;
  size_t off = 0;
  auto take = [&](size_t bytes) -> void* {
    void* p = w + off;
    off += (bytes + 255) & ~(size_t)255;
    return p;
  };
  float* X     = (float*)take((size_t)BB * MP * DD * 4);
  h16*   H16   = (h16*)take((size_t)BB * MP * DD * 2);
  h16*   QKV16 = (h16*)take((size_t)BB * MP * 192 * 2);
  h16*   O16   = (h16*)take((size_t)BB * MP * DD * 2);
  h16*   G16   = (h16*)take((size_t)BB * MP * DD * 2);
  h16*   F16   = (h16*)take((size_t)BB * NN * DD * 2);
  float* Kf    = (float*)take((size_t)BB * NN * DD * 4);
  float* Vf    = (float*)take((size_t)BB * NN * DD * 4);
  h16*   V16   = (h16*)take((size_t)BB * NN * DD * 2);
  float* SLOTS  = (float*)take((size_t)BB * NSLOT * 128 * 4);
  float* SLOTS0 = (float*)take((size_t)BB * NSLOT * 128 * 4);
  float* PI     = (float*)take((size_t)BB * NSLOT * 4);
  (void)ws_size; (void)n_in; (void)in_sizes; (void)out_size;

  float* out0     = (float*)d_out;                 // mlp_out(slots)      32*8*64
  float* out_attn = out0 + (size_t)BB * NSLOT * 64;       // attn        32*8*1024
  float* out2     = out_attn + (size_t)BB * NSLOT * 1024; // mlp_out(slots_init)

  const long totX = (long)BB * MP * DD;
  k_build_x<<<(unsigned)((totX + 255) / 256), 256, 0, stream>>>(inputs, mu_p, ls_p, X);

  const unsigned attn_lds = MP * DH * 2 * 2 + 16 * MP * 4 + 8 * 256 * 4;
  const int ln_rows_full = BB * MP;

  for (int L = 0; L < 2; ++L) {
    k_ln<<<(ln_rows_full + 255) / 256, 256, 0, stream>>>(
        X, (long)MP * DD, 0, MP, t_ln1_g + L * 64, t_ln1_b + L * 64, H16,
        (long)MP * DD, DD);
    k_gemm<0, 0, 0><<<dim3(MT, 12, BB), 32, 0, stream>>>(
        H16, (long)MP * DD, DD, t_Wqkv + (long)L * 192 * 64, DD, nullptr,
        nullptr, 0, 0, QKV16, (long)MP * 192, 192);
    k_attn<<<BB * HEADS, 256, attn_lds, stream>>>(QKV16, O16);
    k_gemm<1, 0, 1><<<dim3(MT, 4, BB), 32, 0, stream>>>(
        O16, (long)MP * DD, DD, t_Wo + (long)L * 64 * 64, DD, t_bo + L * 64,
        X, (long)MP * DD, DD, nullptr, 0, 0);
    k_ln<<<(ln_rows_full + 255) / 256, 256, 0, stream>>>(
        X, (long)MP * DD, 0, MP, t_ln2_g + L * 64, t_ln2_b + L * 64, H16,
        (long)MP * DD, DD);
    k_gemm<1, 1, 0><<<dim3(MT, 4, BB), 32, 0, stream>>>(
        H16, (long)MP * DD, DD, t_W1 + (long)L * 64 * 64, DD, t_b1 + L * 64,
        nullptr, 0, 0, G16, (long)MP * DD, DD);
    k_gemm<1, 0, 1><<<dim3(MT, 4, BB), 32, 0, stream>>>(
        G16, (long)MP * DD, DD, t_W2 + (long)L * 64 * 64, DD, t_b2 + L * 64,
        X, (long)MP * DD, DD, nullptr, 0, 0);
  }

  k_slots_init<<<BB, 256, 0, stream>>>(X, noise, miW1, mib1, miW2, mib2, siW1,
                                       sib1, siW2, sib2, SLOTS, SLOTS0, PI);

  k_ln<<<(BB * NN + 255) / 256, 256, 0, stream>>>(
      X, (long)MP * DD, 2, NN, ln_in_g, ln_in_b, F16, (long)NN * DD, DD);
  k_gemm<0, 0, 0><<<dim3(NN / 16, 4, BB), 32, 0, stream>>>(
      F16, (long)NN * DD, DD, Wk, DD, nullptr, Kf, (long)NN * DD, DD, nullptr, 0, 0);
  k_gemm<0, 0, 0><<<dim3(NN / 16, 4, BB), 32, 0, stream>>>(
      F16, (long)NN * DD, DD, Wv, DD, nullptr, Vf, (long)NN * DD, DD, V16,
      (long)NN * DD, DD);

  k_mlp_out<<<BB * NSLOT, 256, 0, stream>>>(SLOTS0, oW1, ob1, oW2, ob2, out2);

  const unsigned step_lds =
      (NSLOT * 128 * 2 + NSLOT * 64 * 2 + NSLOT * 1024 + NSLOT * 192 * 2 +
       16 * 64 + NSLOT * 64 + NSLOT * 128 + NSLOT * 64 + NSLOT * 2) * 4 +
      16 * 1024 * 2;
  for (int it = 0; it < 4; ++it) {
    k_step<<<BB, 256, step_lds, stream>>>(
        SLOTS, PI, Kf, Vf, V16, Wq, gWih, gWhh, gbih, gbhh, mW1, mb1, mW2, mb2,
        ln_sl_g, ln_sl_b, ln_mu_g, ln_mu_b, out_attn);
  }

  k_mlp_out<<<BB * NSLOT, 256, 0, stream>>>(SLOTS, oW1, ob1, oW2, ob2, out0);
}
